// MTSR_58179626992136
// MI455X (gfx1250) — matrix operand order repaired and checked by reading, not compiled, not run on hardware
//
#include <hip/hip_runtime.h>
#include <math.h>

typedef __attribute__((ext_vector_type(16))) _Float16 v16h;
typedef __attribute__((ext_vector_type(8)))  float    v8f;
typedef __attribute__((ext_vector_type(4)))  float    v4f;

#define DEVINL __device__ __forceinline__

DEVINL float actf(float v, int act) {
  switch (act) {
    case 1: return v > 0.f ? v : 0.01f * v;               // leaky relu
    case 2: return v / (1.f + __expf(-v));                // silu
    case 3: return v > 0.f ? v : 0.f;                     // relu
    case 4: return 1.f / (1.f + __expf(-v));              // sigmoid
    default: return v;
  }
}

DEVINL v8f wmma_f16(const v16h& a, const v16h& b, const v8f& c) {
  return __builtin_amdgcn_wmma_f32_16x16x32_f16(false, a, false, b,
                                                (short)0, c, false, false);
}

// ---------------------------------------------------------------------------
// Generic WMMA GEMM: C[M,N] = act(A[M,K] * B[K,N] (+bias[M]))
// One wave computes one 16x16 C tile.  Main loop: 4x b128 A loads + 16
// offset-gathered B loads, loop-invariant masks, no branches.  Tail chunk
// uses 32-bit clamped offsets (branch-free).
// ---------------------------------------------------------------------------
__global__ __launch_bounds__(32)
void gemm_wmma_kernel(const float* __restrict__ A,
                      const float* __restrict__ Bm,
                      const float* __restrict__ bias,
                      float* __restrict__ C,
                      int M, int N, int K, int lda, int ldb, int ldc,
                      long aB, long bB, long cB,
                      int aMod, int bMod, int cMod,
                      int bTrans, int act) {
  int lane = threadIdx.x & 31;
  int lo = lane & 15;
  int hi = lane >> 4;
  int n0 = blockIdx.x * 16;
  int m0 = blockIdx.y * 16;
  int z  = blockIdx.z;
  const float* Ab = A  + (long)(aMod ? (z % aMod) : z) * aB;
  const float* Bb = Bm + (long)(bMod ? (z % bMod) : z) * bB;
  float*       Cb = C  + (long)(cMod ? (z % cMod) : z) * cB;

  int mA = m0 + lo;
  int nB = n0 + lo;
  bool mok = (mA < M);
  bool nok = (nB < N);
  int sB = bTrans ? 1 : ldb;                       // K-stride within B
  const float* Arow = Ab + (mok ? (long)mA * lda : 0);
  const float* Brow = Bb + (nok ? (long)nB * (long)(bTrans ? ldb : 1) : 0);

  int offB[16];
#pragma unroll
  for (int i = 0; i < 16; ++i) offB[i] = (((i >> 3) << 4) + (i & 7) + (hi << 3)) * sB;

  v8f acc = {};
  int kfull = K & ~31;
  int k0 = 0;
  for (; k0 < kfull; k0 += 32) {
    const float* Ac = Arow + k0 + (hi << 3);
    v4f a0 = *(const v4f*)(Ac);
    v4f a1 = *(const v4f*)(Ac + 4);
    v4f a2 = *(const v4f*)(Ac + 16);
    v4f a3 = *(const v4f*)(Ac + 20);
    const float* Bc = Brow + (long)k0 * sB;
    v16h af, bf;
#pragma unroll
    for (int i = 0; i < 4; ++i) {
      af[i]      = (_Float16)(mok ? a0[i] : 0.f);
      af[i + 4]  = (_Float16)(mok ? a1[i] : 0.f);
      af[i + 8]  = (_Float16)(mok ? a2[i] : 0.f);
      af[i + 12] = (_Float16)(mok ? a3[i] : 0.f);
    }
#pragma unroll
    for (int i = 0; i < 16; ++i) {
      float bv = Bc[offB[i]];
      bf[i] = (_Float16)(nok ? bv : 0.f);
    }
    acc = wmma_f16(af, bf, acc);
  }
  if (k0 < K) {                                    // clamped tail chunk
    int bBase = k0 * sB;
    v16h af, bf;
#pragma unroll
    for (int i = 0; i < 16; ++i) {
      int pa = ((i >> 3) << 4) + (i & 7) + (hi << 3);
      bool va = mok && (k0 + pa < K);
      float av = Arow[va ? (k0 + pa) : 0];
      af[i] = (_Float16)(va ? av : 0.f);
      int pb = ((i >> 3) << 4) + (i & 7) + (hi << 3);
      bool vb = nok && (k0 + pb < K);
      int bi = vb ? (bBase + offB[i]) : 0;
      float bv = Brow[bi];
      bf[i] = (_Float16)(vb ? bv : 0.f);
    }
    acc = wmma_f16(af, bf, acc);
  }
#pragma unroll
  for (int r = 0; r < 8; ++r) {
    int m = m0 + r + (hi << 3);
    int n = n0 + lo;
    if (m < M && n < N) {
      float v = acc[r];
      if (bias) v += bias[m];
      Cb[(long)m * ldc + n] = actf(v, act);
    }
  }
}

// ---------------------------------------------------------------------------
// Implicit-GEMM conv (NCHW, stride 1, square kernel, same-size output).
// K reordered as (tap, channel-chunk): per-tap pixel validity & offsets are
// hoisted; each wave computes a 16x32 tile (one A fragment, two WMMAs).
// All tail indices are 32-bit clamped offsets (branch-free).
// Weights OIHW: element (m, ic, tap) at Wt[m*Cin*KHW + ic*KHW + tap].
// ---------------------------------------------------------------------------
__global__ __launch_bounds__(32)
void conv_wmma_kernel(const float* __restrict__ X,
                      const float* __restrict__ Wt,
                      const float* __restrict__ bias,
                      float* __restrict__ Y,
                      int Cin, int Cout, int H, int W,
                      int KS, int pad, int dil,
                      long xB, long yB, int act) {
  int lane = threadIdx.x & 31;
  int lo = lane & 15;
  int hi = lane >> 4;
  int n0 = blockIdx.x * 32;
  int m0 = blockIdx.y * 16;
  int b  = blockIdx.z;
  const float* Xb = X + (long)b * xB;
  float*       Yb = Y + (long)b * yB;
  int HWo = H * W;
  int KHW = KS * KS;
  long Kfull = (long)Cin * KHW;

  int mA = m0 + lo;
  bool mok = (mA < Cout);
  const float* Arow = Wt + (mok ? (long)mA * Kfull : 0);

  int nB0 = n0 + lo;
  int nB1 = n0 + 16 + lo;
  bool nok0 = nB0 < HWo, nok1 = nB1 < HWo;
  int oy0 = nB0 / W, ox0 = nB0 - oy0 * W;
  int oy1 = nB1 / W, ox1 = nB1 - oy1 * W;

  int offA[16], offB[16];
#pragma unroll
  for (int i = 0; i < 16; ++i) {
    offA[i] = (((i >> 3) << 4) + (i & 7) + (hi << 3)) * KHW;
    offB[i] = (((i >> 3) << 4) + (i & 7) + (hi << 3)) * HWo;
  }

  v8f acc0 = {}, acc1 = {};
  int cfull = Cin & ~31;
  for (int tap = 0; tap < KHW; ++tap) {
    int ky = tap / KS, kx = tap - ky * KS;
    int iy0 = oy0 + dil * ky - pad, ix0 = ox0 + dil * kx - pad;
    int iy1 = oy1 + dil * ky - pad, ix1 = ox1 + dil * kx - pad;
    bool pv0 = nok0 && (iy0 >= 0) && (iy0 < H) && (ix0 >= 0) && (ix0 < W);
    bool pv1 = nok1 && (iy1 >= 0) && (iy1 < H) && (ix1 >= 0) && (ix1 < W);
    const float* B0r = Xb + (pv0 ? (iy0 * W + ix0) : 0);
    const float* B1r = Xb + (pv1 ? (iy1 * W + ix1) : 0);
    const float* Atap = Arow + tap;
    int c0 = 0;
    for (; c0 < cfull; c0 += 32) {
      v16h af, bf0, bf1;
      const float* Ac = Atap + (long)c0 * KHW;
      const float* B0c = B0r + (long)c0 * HWo;
      const float* B1c = B1r + (long)c0 * HWo;
#pragma unroll
      for (int i = 0; i < 16; ++i) {
        float av = Ac[offA[i]];
        af[i] = (_Float16)(mok ? av : 0.f);
        float bv0 = B0c[offB[i]];
        bf0[i] = (_Float16)(pv0 ? bv0 : 0.f);
        float bv1 = B1c[offB[i]];
        bf1[i] = (_Float16)(pv1 ? bv1 : 0.f);
      }
      acc0 = wmma_f16(af, bf0, acc0);
      acc1 = wmma_f16(af, bf1, acc1);
    }
    if (c0 < Cin) {                                // clamped channel tail
      int aBase = c0 * KHW;
      int bBase = c0 * HWo;
      v16h af, bf0, bf1;
#pragma unroll
      for (int i = 0; i < 16; ++i) {
        int pa = ((i >> 3) << 4) + (i & 7) + (hi << 3);
        bool va = mok && (c0 + pa < Cin);
        int ai = va ? (aBase + offA[i]) : 0;
        float av = Atap[ai];
        af[i] = (_Float16)(va ? av : 0.f);
        int pb = ((i >> 3) << 4) + (i & 7) + (hi << 3);
        bool cok = (c0 + pb < Cin);
        bool vb0 = pv0 && cok;
        int bi0 = vb0 ? (bBase + offB[i]) : 0;
        float bv0 = B0r[bi0];
        bf0[i] = (_Float16)(vb0 ? bv0 : 0.f);
        bool vb1 = pv1 && cok;
        int bi1 = vb1 ? (bBase + offB[i]) : 0;
        float bv1 = B1r[bi1];
        bf1[i] = (_Float16)(vb1 ? bv1 : 0.f);
      }
      acc0 = wmma_f16(af, bf0, acc0);
      acc1 = wmma_f16(af, bf1, acc1);
    }
  }
#pragma unroll
  for (int r = 0; r < 8; ++r) {
    int m = m0 + r + (hi << 3);
    if (m < Cout) {
      float bv = bias ? bias[m] : 0.f;
      int na = n0 + lo;
      if (na < HWo) Yb[(long)m * HWo + na] = actf(acc0[r] + bv, act);
      int nb2 = n0 + 16 + lo;
      if (nb2 < HWo) Yb[(long)m * HWo + nb2] = actf(acc1[r] + bv, act);
    }
  }
}

// ---------------------------------------------------------------------------
// LayerNorm across channel dim of NCHW tensor; output NLC (mode 1) or NCHW (0)
// ---------------------------------------------------------------------------
__global__ void ln_nchw_kernel(const float* __restrict__ X,
                               const float* __restrict__ g,
                               const float* __restrict__ bt,
                               float* __restrict__ Y,
                               int C, int HW, long nRows, int outNLC) {
  long idx = (long)blockIdx.x * blockDim.x + threadIdx.x;
  if (idx >= nRows) return;
  long b = idx / HW;
  int  l = (int)(idx - b * HW);
  const float* xb = X + b * (long)C * HW + l;
  float m = 0.f;
  for (int c = 0; c < C; ++c) m += xb[(long)c * HW];
  m /= C;
  float v = 0.f;
  for (int c = 0; c < C; ++c) { float d = xb[(long)c * HW] - m; v += d * d; }
  v /= C;
  float inv = rsqrtf(v + 1e-5f);
  for (int c = 0; c < C; ++c) {
    float o = (xb[(long)c * HW] - m) * inv * g[c] + bt[c];
    if (outNLC) Y[(b * HW + l) * (long)C + c] = o;
    else        Y[(b * (long)C + c) * HW + l] = o;
  }
}

// LayerNorm over last (contiguous) dim of (nRows, C); in-place safe.
__global__ void ln_lastdim_kernel(const float* __restrict__ X,
                                  const float* __restrict__ g,
                                  const float* __restrict__ bt,
                                  float* __restrict__ Y, int C, long nRows) {
  long r = (long)blockIdx.x * blockDim.x + threadIdx.x;
  if (r >= nRows) return;
  const float* xr = X + r * C;
  float m = 0.f;
  for (int c = 0; c < C; ++c) m += xr[c];
  m /= C;
  float v = 0.f;
  for (int c = 0; c < C; ++c) { float d = xr[c] - m; v += d * d; }
  v /= C;
  float inv = rsqrtf(v + 1e-5f);
  float* yr = Y + r * C;
  for (int c = 0; c < C; ++c) yr[c] = (xr[c] - m) * inv * g[c] + bt[c];
}

// Copy a channel slice of NLC tensor into NCHW layout.
__global__ void nlc_slice_to_nchw_kernel(const float* __restrict__ X,
                                         float* __restrict__ Y,
                                         int C, int ldx, int coff, int HW,
                                         long nTot) {
  long idx = (long)blockIdx.x * blockDim.x + threadIdx.x;
  if (idx >= nTot) return;
  long t = idx;
  int l = (int)(t % HW); t /= HW;
  int c = (int)(t % C);
  long b = t / C;
  Y[idx] = X[(b * HW + l) * (long)ldx + coff + c];
}

// Depthwise 3x3 conv (pad 1) + bias + SiLU.
__global__ void dwconv_silu_kernel(const float* __restrict__ X,
                                   const float* __restrict__ Wt,
                                   const float* __restrict__ bs,
                                   float* __restrict__ Y,
                                   int C, int H, int W, long nTot) {
  long idx = (long)blockIdx.x * blockDim.x + threadIdx.x;
  if (idx >= nTot) return;
  int HW = H * W;
  long t = idx;
  int l = (int)(t % HW); t /= HW;
  int c = (int)(t % C);
  long b = t / C;
  int y = l / W, x = l - y * W;
  const float* xb = X + (b * C + c) * (long)HW;
  float acc = bs[c];
  for (int ky = 0; ky < 3; ++ky)
    for (int kx = 0; kx < 3; ++kx) {
      int iy = y + ky - 1, ix = x + kx - 1;
      if (iy >= 0 && iy < H && ix >= 0 && ix < W)
        acc += Wt[c * 9 + ky * 3 + kx] * xb[iy * W + ix];
    }
  Y[idx] = acc / (1.f + __expf(-acc));
}

// Build 4-direction scan inputs xs (B,4,C,L) from NCHW (B,C,H,W).
__global__ void build_xs_kernel(const float* __restrict__ XC,
                                float* __restrict__ XS,
                                int C, int H, int W, long nTot) {
  long idx = (long)blockIdx.x * blockDim.x + threadIdx.x;
  if (idx >= nTot) return;
  int L = H * W;
  long t = idx;
  int l = (int)(t % L); t /= L;
  int c = (int)(t % C); t /= C;
  int k = (int)(t % 4);
  long b = t / 4;
  int src;
  if (k == 0)      src = l;
  else if (k == 1) src = (l % H) * W + (l / H);
  else if (k == 2) src = L - 1 - l;
  else { int lr = L - 1 - l; src = (lr % H) * W + (lr / H); }
  XS[idx] = XC[(b * C + c) * (long)L + src];
}

// delta = softplus(dts @ dt_w^T + dt_b), per (b,k,d,l).
__global__ void delta_kernel(const float* __restrict__ XD,
                             const float* __restrict__ dtw,
                             const float* __restrict__ dtb,
                             float* __restrict__ DL,
                             int C, int L, long nTot) {
  long idx = (long)blockIdx.x * blockDim.x + threadIdx.x;
  if (idx >= nTot) return;
  long t = idx;
  int l = (int)(t % L); t /= L;
  int d = (int)(t % C); t /= C;
  int k = (int)(t % 4);
  long b = t / 4;
  const float* xr = XD + ((b * 4 + k) * 20) * (long)L + l;
  float s = dtb[k * C + d];
  for (int r = 0; r < 4; ++r)
    s += xr[(long)r * L] * dtw[(k * C + d) * 4 + r];
  DL[idx] = (s > 20.f) ? s : log1pf(__expf(s));
}

// Selective scan: one lane per (b,k,c,n) diagonal-state recurrence;
// 8-lane shfl_xor reduction produces y each step. Block = 256 = 32 ch x 8 n.
__global__ void scan_kernel(const float* __restrict__ XS,
                            const float* __restrict__ DL,
                            const float* __restrict__ XD,
                            const float* __restrict__ Alog,
                            const float* __restrict__ Dp,
                            float* __restrict__ YS, int C, int L) {
  int t  = threadIdx.x;
  int n  = t & 7;
  int cl = t >> 3;
  int c  = blockIdx.x * 32 + cl;
  int k  = blockIdx.y;
  int b  = blockIdx.z;
  long baseU = (((long)b * 4 + k) * C + c) * L;
  const float* u  = XS + baseU;
  const float* dl = DL + baseU;
  const float* Bv = XD + (((long)b * 4 + k) * 20 + 4  + n) * (long)L;
  const float* Cv = XD + (((long)b * 4 + k) * 20 + 12 + n) * (long)L;
  float A  = -__expf(Alog[(k * C + c) * 8 + n]);
  float Dd = Dp[k * C + c];
  float* y = YS + baseU;
  float h = 0.f;
  for (int l = 0; l < L; ++l) {
    float dt = dl[l];
    float ut = u[l];
    h = h * __expf(dt * A) + dt * ut * Bv[l];
    float p = h * Cv[l];
    p += __shfl_xor(p, 1, 8);
    p += __shfl_xor(p, 2, 8);
    p += __shfl_xor(p, 4, 8);
    if (n == 0) y[l] = p + Dd * ut;
  }
}

// Merge 4 scan directions back to NLC (B,L,C).
__global__ void merge_kernel(const float* __restrict__ YS,
                             float* __restrict__ Y,
                             int C, int H, int W, long nTot) {
  long idx = (long)blockIdx.x * blockDim.x + threadIdx.x;
  if (idx >= nTot) return;
  int L = H * W;
  long t = idx;
  int l = (int)(t % L); t /= L;
  int c = (int)(t % C);
  long b = t / C;
  int yy = l / W, xx = l - yy * W;
  int s1 = xx * H + yy;
  long base = (b * 4 * (long)C + c) * L;
  long dk = (long)C * L;
  float v = YS[base + l] + YS[base + dk + s1] +
            YS[base + 2 * dk + (L - 1 - l)] + YS[base + 3 * dk + (L - 1 - s1)];
  Y[(b * L + l) * (long)C + c] = v;
}

// y *= silu(z) where z is a channel slice of the NLC xz buffer.
__global__ void gate_kernel(float* __restrict__ Y, const float* __restrict__ XZ,
                            int C, int ldz, int zoff, long nTot) {
  long idx = (long)blockIdx.x * blockDim.x + threadIdx.x;
  if (idx >= nTot) return;
  long r = idx / C;
  int  c = (int)(idx - r * C);
  float z = XZ[r * (long)ldz + zoff + c];
  Y[idx] *= z / (1.f + __expf(-z));
}

__global__ void add_kernel(const float* __restrict__ A,
                           const float* __restrict__ Bp,
                           float* __restrict__ O, long n) {
  long idx = (long)blockIdx.x * blockDim.x + threadIdx.x;
  if (idx < n) O[idx] = A[idx] + Bp[idx];
}

__global__ void add_nlc_to_nchw_kernel(float* __restrict__ CUR,
                                       const float* __restrict__ T,
                                       int C, int HW, long nTot) {
  long idx = (long)blockIdx.x * blockDim.x + threadIdx.x;
  if (idx >= nTot) return;
  long t = idx;
  int l = (int)(t % HW); t /= HW;
  int c = (int)(t % C);
  long b = t / C;
  CUR[idx] += T[(b * HW + l) * (long)C + c];
}

// L2-normalize rows of length L; grid = (channels, batch).
__global__ void l2norm_kernel(const float* __restrict__ X, float* __restrict__ Y,
                              int L, int chTotIn, int chOff, int chTotOut) {
  __shared__ float red[8];
  long ib = ((long)blockIdx.y * chTotIn + chOff + blockIdx.x) * L;
  long ob = ((long)blockIdx.y * chTotOut + blockIdx.x) * L;
  float s = 0.f;
  for (int i = threadIdx.x; i < L; i += blockDim.x) { float v = X[ib + i]; s += v * v; }
  for (int o = 16; o > 0; o >>= 1) s += __shfl_down(s, o, 32);
  if ((threadIdx.x & 31) == 0) red[threadIdx.x >> 5] = s;
  __syncthreads();
  if (threadIdx.x == 0) {
    float tt = 0.f;
    for (int w = 0; w < (int)(blockDim.x >> 5); ++w) tt += red[w];
    red[0] = tt;
  }
  __syncthreads();
  float inv = 1.f / fmaxf(sqrtf(red[0]), 1e-12f);
  for (int i = threadIdx.x; i < L; i += blockDim.x) Y[ob + i] = X[ib + i] * inv;
}

// Softmax over rows of 8 with per-head temperature (rows = B*HEADS*8).
__global__ void softmax8_kernel(float* __restrict__ ATT,
                                const float* __restrict__ temp, long nRows) {
  long idx = (long)blockIdx.x * blockDim.x + threadIdx.x;
  if (idx >= nRows) return;
  int h = (int)((idx >> 3) & 7);
  float* row = ATT + idx * 8;
  float tp = temp[h];
  float mx = -1e30f;
  float v[8];
  for (int j = 0; j < 8; ++j) { v[j] = row[j] * tp; mx = fmaxf(mx, v[j]); }
  float s = 0.f;
  for (int j = 0; j < 8; ++j) { v[j] = __expf(v[j] - mx); s += v[j]; }
  float inv = 1.f / s;
  for (int j = 0; j < 8; ++j) row[j] = v[j] * inv;
}

// Mean over spatial dims per (b,c).
__global__ void spatial_mean_kernel(const float* __restrict__ X,
                                    float* __restrict__ O, int HW, int C) {
  __shared__ float red[8];
  int c = blockIdx.x, b = blockIdx.y;
  long base = ((long)b * C + c) * HW;
  float s = 0.f;
  for (int i = threadIdx.x; i < HW; i += blockDim.x) s += X[base + i];
  for (int o = 16; o > 0; o >>= 1) s += __shfl_down(s, o, 32);
  if ((threadIdx.x & 31) == 0) red[threadIdx.x >> 5] = s;
  __syncthreads();
  if (threadIdx.x == 0) {
    float tt = 0.f;
    for (int w = 0; w < (int)(blockDim.x >> 5); ++w) tt += red[w];
    O[(long)b * C + c] = tt / HW;
  }
}

// Channel-attention MLP (cr == 1): g2 = sigmoid(w2 * relu(w1 . g + b1) + b2).
__global__ void ca_kernel(const float* __restrict__ G,
                          const float* __restrict__ w1, const float* __restrict__ b1,
                          const float* __restrict__ w2, const float* __restrict__ b2,
                          float* __restrict__ G2, int C) {
  __shared__ float hid;
  int b = blockIdx.x, t = threadIdx.x;
  float s = 0.f;
  for (int c = t; c < C; c += 32) s += w1[c] * G[(long)b * C + c];
  for (int o = 16; o > 0; o >>= 1) s += __shfl_down(s, o, 32);
  if (t == 0) hid = fmaxf(s + b1[0], 0.f);
  __syncthreads();
  if (t < C) G2[(long)b * C + t] = 1.f / (1.f + __expf(-(w2[t] * hid + b2[t])));
}

// Channel mean & max per pixel -> (B,2,HW).
__global__ void chan_stats_kernel(const float* __restrict__ X,
                                  float* __restrict__ O, int C, int HW, long nTot) {
  long idx = (long)blockIdx.x * blockDim.x + threadIdx.x;
  if (idx >= nTot) return;
  long b = idx / HW;
  int  l = (int)(idx - b * HW);
  const float* xb = X + b * (long)C * HW + l;
  float m = 0.f, mx = -1e30f;
  for (int c = 0; c < C; ++c) { float v = xb[(long)c * HW]; m += v; mx = fmaxf(mx, v); }
  O[b * 2 * (long)HW + l] = m / C;
  O[b * 2 * (long)HW + HW + l] = mx;
}

// 7x7 spatial-attention conv (2->1, pad 3) + sigmoid.
__global__ void sa_conv_kernel(const float* __restrict__ S,
                               const float* __restrict__ Wt,
                               float* __restrict__ O, int H, int W, long nTot) {
  long idx = (long)blockIdx.x * blockDim.x + threadIdx.x;
  if (idx >= nTot) return;
  int HW = H * W;
  long b = idx / HW;
  int  l = (int)(idx - b * HW);
  int y = l / W, x = l - y * W;
  float acc = 0.f;
  for (int ch = 0; ch < 2; ++ch)
    for (int ky = 0; ky < 7; ++ky)
      for (int kx = 0; kx < 7; ++kx) {
        int iy = y + ky - 3, ix = x + kx - 3;
        if (iy >= 0 && iy < H && ix >= 0 && ix < W)
          acc += Wt[ch * 49 + ky * 7 + kx] * S[(b * 2 + ch) * (long)HW + iy * W + ix];
      }
  O[idx] = 1.f / (1.f + __expf(-acc));
}

// out = g2[b,c]*x4 + sat[b,l]*x4 + x
__global__ void ms_combine_kernel(const float* __restrict__ X4,
                                  const float* __restrict__ G2,
                                  const float* __restrict__ SAT,
                                  const float* __restrict__ Xin,
                                  float* __restrict__ O, int C, int HW, long nTot) {
  long idx = (long)blockIdx.x * blockDim.x + threadIdx.x;
  if (idx >= nTot) return;
  long t = idx;
  int l = (int)(t % HW); t /= HW;
  int c = (int)(t % C);
  long b = t / C;
  float v = X4[idx];
  O[idx] = G2[b * C + c] * v + SAT[b * HW + l] * v + Xin[idx];
}

// Pixel shuffle r=2: in (B,4*Cout,H,W) -> out (B,Cout,2H,2W).
__global__ void pixshuf_kernel(const float* __restrict__ X, float* __restrict__ Y,
                               int Cout, int H, int W, long nTot) {
  long idx = (long)blockIdx.x * blockDim.x + threadIdx.x;
  if (idx >= nTot) return;
  int W2 = W * 2, H2 = H * 2;
  long t = idx;
  int x = (int)(t % W2); t /= W2;
  int y = (int)(t % H2); t /= H2;
  int c = (int)(t % Cout);
  long b = t / Cout;
  int h = y >> 1, ry = y & 1, w = x >> 1, rx = x & 1;
  Y[idx] = X[(((b * (long)Cout * 4) + (c * 4 + ry * 2 + rx)) * H + h) * (long)W + w];
}

// ===========================================================================
extern "C" void kernel_launch(void* const* d_in, const int* in_sizes, int n_in,
                              void* d_out, int out_size, void* d_ws, size_t ws_size,
                              hipStream_t stream) {
  (void)in_sizes; (void)n_in; (void)out_size; (void)ws_size;
  const float* x   = (const float*)d_in[0];
  const float* lms = (const float*)d_in[1];
  auto P = [&](int i) { return (const float*)d_in[i]; };

  const int Bn = 4, F = 64, Hs = 64, Wsz = 64;
  const int HW = Hs * Wsz, L = HW;
  float* ws = (float*)d_ws;
  size_t off = 0;
  auto alloc = [&](size_t n) { float* p = ws + off; off += n; return p; };
  float* CAT = alloc((size_t)Bn * 93 * HW);
  float* X4  = alloc((size_t)Bn * 31 * HW);
  float* GB  = alloc((size_t)Bn * 31);
  float* G2  = alloc((size_t)Bn * 31);
  float* SST = alloc((size_t)Bn * 2 * HW);
  float* SAT = alloc((size_t)Bn * HW);
  float* MSO = alloc((size_t)Bn * 31 * HW);
  float* SH  = alloc((size_t)Bn * F * HW);
  float* CUR = alloc((size_t)Bn * F * HW);
  float* TA  = alloc((size_t)Bn * L * F);
  float* TB  = alloc((size_t)Bn * L * F);
  float* XZ  = alloc((size_t)Bn * L * 2 * F);
  float* XS  = alloc((size_t)Bn * 4 * F * L);
  float* XD  = alloc((size_t)Bn * 4 * 20 * L);
  float* DLb = alloc((size_t)Bn * 4 * F * L);
  float* YSb = alloc((size_t)Bn * 4 * F * L);
  float* XM  = alloc((size_t)Bn * F * HW);
  float* XC  = alloc((size_t)Bn * F * HW);
  float* G1  = alloc((size_t)Bn * L * F);
  float* QKV = alloc((size_t)Bn * 192 * HW);
  float* QN  = alloc((size_t)Bn * F * HW);
  float* KN  = alloc((size_t)Bn * F * HW);
  float* ATT = alloc((size_t)Bn * 8 * 64);
  float* TRO = alloc((size_t)Bn * F * HW);
  float* TRP = alloc((size_t)Bn * F * HW);
  float* UP0 = alloc((size_t)Bn * F * HW);
  float* UC1 = alloc((size_t)Bn * 256 * HW);
  float* UP1 = alloc((size_t)Bn * F * HW * 4);
  float* UC2 = alloc((size_t)Bn * 256 * HW * 4);
  float* UP2 = alloc((size_t)Bn * F * HW * 16);
  float* SKP = alloc((size_t)Bn * F * HW * 16);

  auto g1 = [](long n) { return dim3((unsigned)((n + 255) / 256)); };

  auto conv = [&](const float* Xp, const float* Wp, const float* bp, float* Yp,
                  int Cin, int Cout, int Hh, int Ww, int KS, int pad, int dil,
                  long xB, long yB, int act) {
    dim3 g((Hh * Ww + 31) / 32, (Cout + 15) / 16, Bn);
    conv_wmma_kernel<<<g, dim3(32), 0, stream>>>(Xp, Wp, bp, Yp, Cin, Cout, Hh, Ww,
                                                 KS, pad, dil, xB, yB, act);
  };
  auto gemm = [&](const float* Ap, const float* Bp, const float* bp, float* Cp,
                  int M, int N, int K, int lda, int ldb, int ldc,
                  long aB, long bB, long cB, int aMod, int bMod, int cMod,
                  int nz, int bTrans, int act) {
    dim3 g((N + 15) / 16, (M + 15) / 16, nz);
    gemm_wmma_kernel<<<g, dim3(32), 0, stream>>>(Ap, Bp, bp, Cp, M, N, K, lda, ldb,
                                                 ldc, aB, bB, cB, aMod, bMod, cMod,
                                                 bTrans, act);
  };

  // ---------------- multiscale ----------------
  conv(x, P(2), P(3), CAT,             31, 31, Hs, Wsz, 3, 1, 1, (long)31*HW, (long)93*HW, 1);
  conv(x, P(4), P(5), CAT + 31*HW,     31, 31, Hs, Wsz, 3, 3, 3, (long)31*HW, (long)93*HW, 1);
  conv(x, P(6), P(7), CAT + 62*HW,     31, 31, Hs, Wsz, 3, 5, 5, (long)31*HW, (long)93*HW, 1);
  conv(CAT, P(8), P(9), X4,            93, 31, Hs, Wsz, 1, 0, 1, (long)93*HW, (long)31*HW, 0);
  spatial_mean_kernel<<<dim3(31, Bn), 256, 0, stream>>>(X4, GB, HW, 31);
  ca_kernel<<<Bn, 32, 0, stream>>>(GB, P(10), P(11), P(12), P(13), G2, 31);
  chan_stats_kernel<<<g1((long)Bn*HW), 256, 0, stream>>>(X4, SST, 31, HW, (long)Bn*HW);
  sa_conv_kernel<<<g1((long)Bn*HW), 256, 0, stream>>>(SST, P(14), SAT, Hs, Wsz, (long)Bn*HW);
  ms_combine_kernel<<<g1((long)Bn*31*HW), 256, 0, stream>>>(X4, G2, SAT, x, MSO, 31, HW,
                                                            (long)Bn*31*HW);
  // ---------------- shallow ----------------
  conv(MSO, P(15), P(16), SH, 31, 64, Hs, Wsz, 3, 1, 1, (long)31*HW, (long)64*HW, 0);
  hipMemcpyAsync(CUR, SH, (size_t)Bn*F*HW*sizeof(float), hipMemcpyDeviceToDevice, stream);

  // ---------------- SS2D block ----------------
  auto ss2d = [&](const float* in, float* out, int pb) {
    gemm(in, P(pb + 0), nullptr, XZ, L, 128, 64, 64, 64, 128,
         (long)L*64, 0, (long)L*128, 0, 0, 0, Bn, 1, 0);
    nlc_slice_to_nchw_kernel<<<g1((long)Bn*F*HW), 256, 0, stream>>>(XZ, XM, 64, 128, 0, HW,
                                                                    (long)Bn*F*HW);
    dwconv_silu_kernel<<<g1((long)Bn*F*HW), 256, 0, stream>>>(XM, P(pb+1), P(pb+2), XC,
                                                              64, Hs, Wsz, (long)Bn*F*HW);
    build_xs_kernel<<<g1((long)Bn*4*F*L), 256, 0, stream>>>(XC, XS, 64, Hs, Wsz,
                                                            (long)Bn*4*F*L);
    gemm(P(pb + 3), XS, nullptr, XD, 20, L, 64, 64, L, L,
         (long)20*64, (long)64*L, (long)20*L, 4, 0, 0, Bn*4, 0, 0);
    delta_kernel<<<g1((long)Bn*4*F*L), 256, 0, stream>>>(XD, P(pb+4), P(pb+5), DLb,
                                                         64, L, (long)Bn*4*F*L);
    scan_kernel<<<dim3(2, 4, Bn), 256, 0, stream>>>(XS, DLb, XD, P(pb+6), P(pb+7), YSb,
                                                    64, L);
    merge_kernel<<<g1((long)Bn*F*L), 256, 0, stream>>>(YSb, G1, 64, Hs, Wsz, (long)Bn*F*L);
    ln_lastdim_kernel<<<g1((long)Bn*L), 256, 0, stream>>>(G1, P(pb+8), P(pb+9), G1, 64,
                                                          (long)Bn*L);
    gate_kernel<<<g1((long)Bn*L*F), 256, 0, stream>>>(G1, XZ, 64, 128, 64, (long)Bn*L*F);
    gemm(G1, P(pb + 10), nullptr, out, L, 64, 64, 64, 64, 64,
         (long)L*64, 0, (long)L*64, 0, 0, 0, Bn, 1, 0);
  };

  // ---------------- transformer ----------------
  auto transformer = [&](const float* Xin, float* Yout, int tb) {
    conv(Xin, P(tb + 1), nullptr, QKV, 64, 192, Hs, Wsz, 1, 0, 1,
         (long)64*HW, (long)192*HW, 0);
    l2norm_kernel<<<dim3(64, Bn), 256, 0, stream>>>(QKV, QN, L, 192, 0, 64);
    l2norm_kernel<<<dim3(64, Bn), 256, 0, stream>>>(QKV, KN, L, 192, 64, 64);
    gemm(QN, KN, nullptr, ATT, 8, 8, L, L, L, 8,
         (long)8*L, (long)8*L, 64, 0, 0, 0, Bn*8, 1, 0);
    softmax8_kernel<<<1, 256, 0, stream>>>(ATT, P(tb), (long)Bn*64);
    for (int b = 0; b < Bn; ++b)
      gemm(ATT + b*8*64, QKV + ((long)b*192 + 128)*L, nullptr, TRO + (long)b*64*L,
           8, L, 8, 8, L, L, 64, (long)8*L, (long)8*L, 0, 0, 0, 8, 0, 0);
    conv(TRO, P(tb + 2), nullptr, Yout, 64, 64, Hs, Wsz, 1, 0, 1,
         (long)64*HW, (long)64*HW, 0);
  };

  // ---------------- 4 MTB stages ----------------
  long nF = (long)Bn * F * HW;
  for (int s = 0; s < 4; ++s) {
    int base = 17 + s * 29;
    ln_nchw_kernel<<<g1((long)Bn*HW), 256, 0, stream>>>(CUR, P(base), P(base+1), TA,
                                                        64, HW, (long)Bn*HW, 1);
    ss2d(TA, TB, base + 4);
    ss2d(TB, TA, base + 15);
    add_nlc_to_nchw_kernel<<<g1(nF), 256, 0, stream>>>(CUR, TA, 64, HW, nF);
    ln_nchw_kernel<<<g1((long)Bn*HW), 256, 0, stream>>>(CUR, P(base+2), P(base+3), XM,
                                                        64, HW, (long)Bn*HW, 0);
    transformer(XM, TRP, base + 26);
    add_kernel<<<g1(nF), 256, 0, stream>>>(CUR, TRP, CUR, nF);
  }

  // ---------------- upsample + skip + tail ----------------
  add_kernel<<<g1(nF), 256, 0, stream>>>(CUR, SH, UP0, nF);
  conv(UP0, P(133), P(134), UC1, 64, 256, 64, 64, 3, 1, 1, (long)64*HW, (long)256*HW, 0);
  pixshuf_kernel<<<g1((long)Bn*64*HW*4), 256, 0, stream>>>(UC1, UP1, 64, 64, 64,
                                                           (long)Bn*64*HW*4);
  conv(UP1, P(135), P(136), UC2, 64, 256, 128, 128, 3, 1, 1,
       (long)64*HW*4, (long)256*HW*4, 0);
  pixshuf_kernel<<<g1((long)Bn*64*HW*16), 256, 0, stream>>>(UC2, UP2, 64, 128, 128,
                                                            (long)Bn*64*HW*16);
  conv(lms, P(137), P(138), SKP, 31, 64, 256, 256, 3, 1, 1,
       (long)31*HW*16, (long)64*HW*16, 0);
  add_kernel<<<g1((long)Bn*64*HW*16), 256, 0, stream>>>(UP2, SKP, UP2, (long)Bn*64*HW*16);
  conv(UP2, P(139), P(140), (float*)d_out, 64, 31, 256, 256, 3, 1, 1,
       (long)64*HW*16, (long)31*HW*16, 0);
}
